// GCN_PyG_46488726012208
// MI455X (gfx1250) — matrix operand order repaired and checked by reading, not compiled, not run on hardware
//
#include <hip/hip_runtime.h>
#include <hip/hip_bf16.h>

typedef __attribute__((ext_vector_type(16))) _Float16 v16h;
typedef __attribute__((ext_vector_type(8)))  _Float16 v8h;
typedef __attribute__((ext_vector_type(4)))  _Float16 v4h;
typedef __attribute__((ext_vector_type(8)))  float    v8f;

#define DEV_INLINE __device__ __forceinline__

DEV_INLINE void atomAddF(float* p, float v) {
    // native global_atomic_add_f32 (no return -> STOREcnt, no CAS loop)
    unsafeAtomicAdd(p, v);
}

// ---------------------------------------------------------------- utilities
__global__ void k_zero_f32(float* __restrict__ p, int n) {
    int i = blockIdx.x * blockDim.x + threadIdx.x;
    int stride = gridDim.x * blockDim.x;
    for (; i < n; i += stride) p[i] = 0.0f;
}

__global__ void k_deg(const int* __restrict__ dst, float* __restrict__ deg, int E) {
    int e = blockIdx.x * blockDim.x + threadIdx.x;
    if (e < E) atomAddF(&deg[dst[e]], 1.0f);
}

__global__ void k_rsqrt_inplace(float* __restrict__ d, int n) {
    int i = blockIdx.x * blockDim.x + threadIdx.x;
    if (i < n) d[i] = rsqrtf(d[i] + 1.0f);   // deg includes self-loop
}

__global__ void k_norm(const int* __restrict__ src, const int* __restrict__ dst,
                       const float* __restrict__ dis, float* __restrict__ norm, int E) {
    int e = blockIdx.x * blockDim.x + threadIdx.x;
    if (e < E) norm[e] = dis[src[e]] * dis[dst[e]];
}

__global__ void k_f32_to_f16(const float* __restrict__ in, _Float16* __restrict__ out, int n4) {
    int i = blockIdx.x * blockDim.x + threadIdx.x;
    if (i >= n4) return;
    const float4 v = ((const float4*)in)[i];
    v4h h; h.x = (_Float16)v.x; h.y = (_Float16)v.y; h.z = (_Float16)v.z; h.w = (_Float16)v.w;
    ((v4h*)out)[i] = h;
}

// WT[n*K + k] = (f16) W[k*N + n]   (W row-major [K,N]); tiny matrix, L2-resident
__global__ void k_transposeW(const float* __restrict__ W, _Float16* __restrict__ WT,
                             int K, int N) {
    int t = blockIdx.x * blockDim.x + threadIdx.x;
    if (t >= K * N) return;
    int k = t / N, n = t - k * N;
    WT[(long)n * K + k] = (_Float16)W[t];
}

// ---------------------------------------------------------------- WMMA GEMM
DEV_INLINE v8f wmma_f16(v16h a, v16h b, v8f c) {
    return __builtin_amdgcn_wmma_f32_16x16x32_f16(
        /*neg_a=*/false, a, /*neg_b=*/false, b,
        /*c_mod=*/(short)0, c, /*reuse_a=*/false, /*reuse_b=*/false);
}

// C[M,N] = A[M,K] @ WT[N,K]^T   (A f16 row-major, WT f16 row-major = W^T)
// 8 waves / block; wave computes a 16(M) x 64(N) tile via 4 accumulators.
__global__ void __launch_bounds__(256)
k_gemm_wmma(const _Float16* __restrict__ A, const _Float16* __restrict__ WT,
            float* __restrict__ C, int M, int K, int N) {
    const int lane  = threadIdx.x & 31;
    const int wave  = threadIdx.x >> 5;
    const int strip = blockIdx.x * 8 + wave;        // 16-row strip index
    if (strip * 16 >= M) return;                    // wave-uniform exit
    const int nBase = blockIdx.y * 64;

    const int l15  = lane & 15;
    const int hi   = lane >> 4;                     // lane half
    const int aoff = hi * 16;                       // A: k = hi*16 + {0..15}
    const int boff = hi * 16;                       // B: k = hi*16 + {0..15}

    const _Float16* __restrict__ arow = A  + (long)(strip * 16 + l15) * K;
    const _Float16* __restrict__ b0   = WT + (long)(nBase +  0 + l15) * K + boff;
    const _Float16* __restrict__ b1   = WT + (long)(nBase + 16 + l15) * K + boff;
    const _Float16* __restrict__ b2   = WT + (long)(nBase + 32 + l15) * K + boff;
    const _Float16* __restrict__ b3   = WT + (long)(nBase + 48 + l15) * K + boff;

    v8f acc0 = {}, acc1 = {}, acc2 = {}, acc3 = {};

    #pragma unroll 2
    for (int kt = 0; kt < K; kt += 32) {
        v8h alo = *(const v8h*)(arow + kt + aoff);
        v8h ahi = *(const v8h*)(arow + kt + aoff + 8);
        v16h a  = __builtin_shufflevector(alo, ahi,
                    0,1,2,3,4,5,6,7,8,9,10,11,12,13,14,15);
        acc0 = wmma_f16(a, *(const v16h*)(b0 + kt), acc0);
        acc1 = wmma_f16(a, *(const v16h*)(b1 + kt), acc1);
        acc2 = wmma_f16(a, *(const v16h*)(b2 + kt), acc2);
        acc3 = wmma_f16(a, *(const v16h*)(b3 + kt), acc3);
    }

    // C/D layout: VGPR r -> m = r + 8*hi, n = lane&15
    const int mBase = strip * 16 + hi * 8;
    #pragma unroll
    for (int r = 0; r < 8; ++r) {
        float* crow = C + (long)(mBase + r) * N + nBase + l15;
        crow[0]  = acc0[r];
        crow[16] = acc1[r];
        crow[32] = acc2[r];
        crow[48] = acc3[r];
    }
}

// ---------------------------------------------------------------- scatter path
// agg[i,:] = h[i,:] * dis[i]^2   (self-loop term)
__global__ void k_agg_init(const float* __restrict__ h, const float* __restrict__ dis,
                           float* __restrict__ agg, int n4, int N) {
    int i = blockIdx.x * blockDim.x + threadIdx.x;
    if (i >= n4) return;
    int row = (i << 2) / N;
    float d = dis[row]; d *= d;
    float4 v = ((const float4*)h)[i];
    v.x *= d; v.y *= d; v.z *= d; v.w *= d;
    ((float4*)agg)[i] = v;
}

// agg[dst[e],:] += h[src[e],:] * norm[e]  -- L2-resident atomics
__global__ void k_edge_scatter(const int* __restrict__ src, const int* __restrict__ dst,
                               const float* __restrict__ norm, const float* __restrict__ h,
                               float* __restrict__ agg, int E, int N) {
    const int vpr = N >> 2;                          // float4 per row
    int t = blockIdx.x * blockDim.x + threadIdx.x;
    if (t >= E * vpr) return;
    int e  = t / vpr;
    int f  = (t - e * vpr) << 2;
    int s  = src[e], d = dst[e];
    float nm = norm[e];
    float4 hv = *(const float4*)(h + (long)s * N + f);
    float* out = agg + (long)d * N + f;
    atomAddF(out + 0, hv.x * nm);
    atomAddF(out + 1, hv.y * nm);
    atomAddF(out + 2, hv.z * nm);
    atomAddF(out + 3, hv.w * nm);
}

// out16 = (f16) relu(agg + b)   (next-layer GEMM input)
__global__ void k_bias_relu_f16(const float* __restrict__ agg, const float* __restrict__ b,
                                _Float16* __restrict__ out16, int n4, int N) {
    int i = blockIdx.x * blockDim.x + threadIdx.x;
    if (i >= n4) return;
    int f = (i << 2) % N;
    float4 v = ((const float4*)agg)[i];
    v.x = fmaxf(v.x + b[f + 0], 0.0f);
    v.y = fmaxf(v.y + b[f + 1], 0.0f);
    v.z = fmaxf(v.z + b[f + 2], 0.0f);
    v.w = fmaxf(v.w + b[f + 3], 0.0f);
    v4h h; h.x = (_Float16)v.x; h.y = (_Float16)v.y; h.z = (_Float16)v.z; h.w = (_Float16)v.w;
    ((v4h*)out16)[i] = h;
}

// layer 3: accum += sum(relu(agg + b))   (fused global reduction)
__global__ void k_bias_relu_reduce(const float* __restrict__ agg, const float* __restrict__ b,
                                   float* __restrict__ accum, int n4, int N) {
    __shared__ float sm[256];
    int i = blockIdx.x * blockDim.x + threadIdx.x;
    int stride = gridDim.x * blockDim.x;
    float s = 0.0f;
    for (; i < n4; i += stride) {
        int f = (i << 2) % N;
        float4 v = ((const float4*)agg)[i];
        s += fmaxf(v.x + b[f + 0], 0.0f);
        s += fmaxf(v.y + b[f + 1], 0.0f);
        s += fmaxf(v.z + b[f + 2], 0.0f);
        s += fmaxf(v.w + b[f + 3], 0.0f);
    }
    sm[threadIdx.x] = s;
    __syncthreads();
    for (int off = 128; off > 0; off >>= 1) {
        if (threadIdx.x < off) sm[threadIdx.x] += sm[threadIdx.x + off];
        __syncthreads();
    }
    if (threadIdx.x == 0) atomAddF(accum, sm[0]);
}

__global__ void k_finalize(const float* __restrict__ accum, float* __restrict__ out) {
    out[0] = accum[0] * (1.0f / 64.0f);
}

// ---------------------------------------------------------------- launch
extern "C" void kernel_launch(void* const* d_in, const int* in_sizes, int n_in,
                              void* d_out, int out_size, void* d_ws, size_t ws_size,
                              hipStream_t stream) {
    const float* x[3]  = { (const float*)d_in[0], (const float*)d_in[1], (const float*)d_in[2] };
    const int*   ei[3] = { (const int*)d_in[3], (const int*)d_in[4], (const int*)d_in[5] };
    const float* W[3]  = { (const float*)d_in[6], (const float*)d_in[8], (const float*)d_in[10] };
    const float* bb[3] = { (const float*)d_in[7], (const float*)d_in[9], (const float*)d_in[11] };

    const int Din = 256;
    const int Nn  = in_sizes[0] / Din;        // 50000 (multiple of 16)
    const int E   = in_sizes[3] / 2;          // 800000
    const int Kd[3] = { 256, 256, 128 };
    const int Nd[3] = { 256, 128,  64 };

    // workspace carve-out (256B aligned slabs)
    char* base = (char*)d_ws;
    size_t off = 0;
    auto carve = [&](size_t bytes) -> char* {
        char* p = base + off;
        off = (off + bytes + 255) & ~(size_t)255;
        return p;
    };
    float*    dis   = (float*)   carve((size_t)Nn * 4);
    float*    norm  = (float*)   carve((size_t)E * 4);
    _Float16* x16   = (_Float16*)carve((size_t)Nn * 256 * 2);
    float*    hbuf  = (float*)   carve((size_t)Nn * 256 * 4);
    float*    agg   = (float*)   carve((size_t)Nn * 256 * 4);
    _Float16* wt16  = (_Float16*)carve((size_t)256 * 256 * 2);
    float*    accum = (float*)   carve(256);
    if (off > ws_size) return;                // workspace too small: bail safely

    const int TB = 256;
    auto cdiv = [](int a, int b) { return (a + b - 1) / b; };

    k_zero_f32<<<1, 32, 0, stream>>>(accum, 1);

    for (int g = 0; g < 3; ++g) {
        const int* src = ei[g];
        const int* dst = ei[g] + E;

        // degrees -> dis -> per-edge norm (shared by all 3 layers)
        k_zero_f32<<<cdiv(Nn, TB), TB, 0, stream>>>(dis, Nn);
        k_deg<<<cdiv(E, TB), TB, 0, stream>>>(dst, dis, E);
        k_rsqrt_inplace<<<cdiv(Nn, TB), TB, 0, stream>>>(dis, Nn);
        k_norm<<<cdiv(E, TB), TB, 0, stream>>>(src, dst, dis, norm, E);

        // f16 activations for layer-1 GEMM
        k_f32_to_f16<<<cdiv(Nn * Din / 4, TB), TB, 0, stream>>>(x[g], x16, Nn * Din / 4);

        for (int l = 0; l < 3; ++l) {
            const int K = Kd[l], N = Nd[l];
            k_transposeW<<<cdiv(K * N, TB), TB, 0, stream>>>(W[l], wt16, K, N);

            dim3 gg(cdiv(Nn / 16, 8), N / 64);
            k_gemm_wmma<<<gg, TB, 0, stream>>>(x16, wt16, hbuf, Nn, K, N);

            const int n4 = Nn * N / 4;
            k_agg_init<<<cdiv(n4, TB), TB, 0, stream>>>(hbuf, dis, agg, n4, N);
            k_edge_scatter<<<cdiv(E * (N / 4), TB), TB, 0, stream>>>(
                src, dst, norm, hbuf, agg, E, N);

            if (l < 2) {
                k_bias_relu_f16<<<cdiv(n4, TB), TB, 0, stream>>>(agg, bb[l], x16, n4, N);
            } else {
                k_bias_relu_reduce<<<1024, TB, 0, stream>>>(agg, bb[l], accum, n4, N);
            }
        }
    }
    k_finalize<<<1, 1, 0, stream>>>(accum, (float*)d_out);
}
